// ExpertChoiceMoELayer_33586644254993
// MI455X (gfx1250) — matrix operand order repaired and checked by reading, not compiled, not run on hardware
//
#include <hip/hip_runtime.h>
#include <hip/hip_bf16.h>
#include <math.h>

// Problem constants (from reference): B=4,S=4096 -> N=16384 tokens
#define N_TOK 16384
#define H_DIM 1024
#define E_NUM 8
#define CAP   2048
#define I_DIM 2048

typedef __bf16 bf16;
typedef __attribute__((ext_vector_type(16))) bf16  v16bf;
typedef __attribute__((ext_vector_type(8)))  bf16  v8bf;
typedef __attribute__((ext_vector_type(8)))  float v8f;
typedef __attribute__((ext_vector_type(4)))  unsigned int u32x4;
typedef __attribute__((ext_vector_type(8)))  int i32x8;
typedef __attribute__((ext_vector_type(4)))  int i32x4;

#if __has_builtin(__builtin_amdgcn_tensor_load_to_lds)
#define HAVE_TDM 1
#endif

#if __has_builtin(__builtin_amdgcn_s_wait_asynccnt)
#define WAIT_ASYNC(n) __builtin_amdgcn_s_wait_asynccnt(n)
#else
#define WAIT_ASYNC(n) asm volatile("s_wait_asynccnt %0" :: "i"(n) : "memory")
#endif

#if __has_builtin(__builtin_amdgcn_s_wait_tensorcnt)
#define WAIT_TENSOR_N(n) __builtin_amdgcn_s_wait_tensorcnt(n)
#else
#define WAIT_TENSOR_N(n) asm volatile("s_wait_tensorcnt %0" :: "i"(n) : "memory")
#endif

__device__ __forceinline__ float safe_clampf(float x, float m) {
    if (__builtin_isnan(x)) return 0.0f;
    x = x > m ? m : x;
    x = x < -m ? -m : x;
    return x;
}

__device__ __forceinline__ float siluf(float x) {
    return x / (1.0f + __expf(-x));
}

// LDS byte address of a generic pointer that points into LDS
__device__ __forceinline__ unsigned lds_off(const void* p) {
    return (unsigned)(unsigned long long)(__attribute__((address_space(3))) const void*)p;
}

// ASYNCcnt-tracked global -> LDS 128-bit copy (one instruction per wave, 16B per lane)
__device__ __forceinline__ void async_b128(unsigned lds, const bf16* g) {
    asm volatile("global_load_async_to_lds_b128 %0, %1, off"
                 :: "v"(lds), "v"((unsigned long long)g) : "memory");
}

#if defined(HAVE_TDM)
// TDM: stage a 2D tile (tile_d1 rows x tile_d0 elems, bf16) of a row-major tensor
// whose fastest dim has stride stride_d0 (elements). gaddr points at the TILE start.
__device__ __forceinline__ void tdm_load_2d(unsigned lds_byte_addr, const void* gaddr,
                                            int tile_d0, int tile_d1,
                                            int tensor_d0, int tensor_d1, int stride_d0) {
    unsigned long long ga = (unsigned long long)gaddr;
    u32x4 g0;
    g0[0] = 1u;                                              // count=1 valid descriptor
    g0[1] = lds_byte_addr;                                   // lds_addr [63:32]
    g0[2] = (unsigned)(ga & 0xFFFFFFFFu);                    // global_addr lo
    g0[3] = (unsigned)((ga >> 32) & 0x1FFFFFFu) | (2u << 30); // global_addr hi | type=2
    i32x8 g1;
    g1[0] = (1 << 16);                                       // workgroup_mask=0, data_size=1 (2B)
    g1[1] = (tensor_d0 & 0xFFFF) << 16;                      // atomic_barrier=0 | tensor_dim0 lo16
    g1[2] = ((unsigned)tensor_d0 >> 16) | ((tensor_d1 & 0xFFFF) << 16);
    g1[3] = ((unsigned)tensor_d1 >> 16) | (tile_d0 << 16);   // tensor_dim1 hi | tile_dim0
    g1[4] = (tile_d1 & 0xFFFF);                              // tile_dim1 | tile_dim2=0
    g1[5] = stride_d0;                                       // tensor_dim0_stride lo32
    g1[6] = 0;                                               // stride hi | dim1_stride lo
    g1[7] = 0;
    i32x4 z4 = {0, 0, 0, 0};
#if __clang_major__ >= 23
    i32x8 z8 = {0, 0, 0, 0, 0, 0, 0, 0};
    __builtin_amdgcn_tensor_load_to_lds(g0, g1, z4, z4, z8, 0);
#else
    __builtin_amdgcn_tensor_load_to_lds(g0, g1, z4, z4, 0);
#endif
}
#endif

// ---------------- zero ----------------
__global__ void k_zero(float* p, int n) {
    int i = blockIdx.x * blockDim.x + threadIdx.x;
    int st = gridDim.x * blockDim.x;
    for (; i < n; i += st) p[i] = 0.0f;
}

// ---------------- block reduce helper ----------------
__device__ __forceinline__ float block_reduce_sum_256(float v, float* red) {
    int t = threadIdx.x;
    red[t] = v; __syncthreads();
    for (int s = 128; s > 0; s >>= 1) {
        if (t < s) red[t] += red[t + s];
        __syncthreads();
    }
    float r = red[0]; __syncthreads();
    return r;
}

// ---------------- LayerNorm + gate logits + lse^2 ----------------
__global__ __launch_bounds__(256)
void k_ln_gate(const float* __restrict__ x, const float* __restrict__ gw,
               float* __restrict__ logits, float* __restrict__ lse2) {
    __shared__ float red[256];
    __shared__ float slog[E_NUM];
    int t = blockIdx.x, tid = threadIdx.x;

    float xv[4];
    #pragma unroll
    for (int j = 0; j < 4; ++j) {
        int h = tid + j * 256;
        xv[j] = safe_clampf(x[(size_t)t * H_DIM + h], 1000.0f);
    }
    float s = xv[0] + xv[1] + xv[2] + xv[3];
    s = block_reduce_sum_256(s, red);
    float mu = s * (1.0f / H_DIM);
    float vs = 0.0f;
    #pragma unroll
    for (int j = 0; j < 4; ++j) { float d = xv[j] - mu; vs += d * d; }
    vs = block_reduce_sum_256(vs, red);
    float rstd = rsqrtf(vs * (1.0f / H_DIM) + 1e-5f);

    float hn[4];
    #pragma unroll
    for (int j = 0; j < 4; ++j)
        hn[j] = safe_clampf((xv[j] - mu) * rstd, 100.0f);

    for (int e = 0; e < E_NUM; ++e) {
        float p = 0.0f;
        #pragma unroll
        for (int j = 0; j < 4; ++j)
            p += hn[j] * gw[(size_t)e * H_DIM + tid + j * 256];
        p = block_reduce_sum_256(p, red);
        if (tid == 0) slog[e] = fminf(fmaxf(p, -20.0f), 20.0f);
    }
    __syncthreads();
    if (tid < E_NUM) logits[(size_t)t * E_NUM + tid] = slog[tid];
    if (tid == 0) {
        float mx = -1e30f;
        for (int e = 0; e < E_NUM; ++e) mx = fmaxf(mx, slog[e]);
        float se = 0.0f;
        for (int e = 0; e < E_NUM; ++e) se += __expf(slog[e] - mx);
        float lse = mx + __logf(se);
        atomicAdd(lse2, lse * lse);
    }
}

// ---------------- softmax over tokens (axis=0), one block per expert ----------------
__global__ __launch_bounds__(1024)
void k_softmax(const float* __restrict__ logits, float* __restrict__ probs) {
    __shared__ float red[1024];
    int e = blockIdx.x, tid = threadIdx.x, nt = blockDim.x;

    float m = -1e30f;
    for (int i = tid; i < N_TOK; i += nt) m = fmaxf(m, logits[(size_t)i * E_NUM + e]);
    red[tid] = m; __syncthreads();
    for (int s = nt >> 1; s > 0; s >>= 1) {
        if (tid < s) red[tid] = fmaxf(red[tid], red[tid + s]);
        __syncthreads();
    }
    m = red[0]; __syncthreads();

    float sum = 0.0f;
    for (int i = tid; i < N_TOK; i += nt) sum += __expf(logits[(size_t)i * E_NUM + e] - m);
    red[tid] = sum; __syncthreads();
    for (int s = nt >> 1; s > 0; s >>= 1) {
        if (tid < s) red[tid] += red[tid + s];
        __syncthreads();
    }
    sum = red[0]; __syncthreads();

    float inv = 1.0f / sum;
    for (int i = tid; i < N_TOK; i += nt) {
        float p = __expf(logits[(size_t)i * E_NUM + e] - m) * inv;
        probs[(size_t)i * E_NUM + e] = fminf(fmaxf(p, 1e-5f), 1.0f);
    }
}

// ---------------- top-CAP selection per expert via histogram threshold ----------------
#define NBINS 4096
__global__ __launch_bounds__(256)
void k_topk(const float* __restrict__ probs, int* __restrict__ tIdx, float* __restrict__ tPrb) {
    __shared__ int hist[NBINS];
    __shared__ int s_bin, s_above, ctrA, ctrT;
    int e = blockIdx.x, tid = threadIdx.x;

    for (int i = tid; i < NBINS; i += 256) hist[i] = 0;
    __syncthreads();

    for (int i = tid; i < N_TOK; i += 256) {
        unsigned b = __float_as_uint(probs[(size_t)i * E_NUM + e]) >> 19;
        if (b > NBINS - 1) b = NBINS - 1;
        atomicAdd(&hist[b], 1);
    }
    __syncthreads();

    if (tid == 0) {
        int cum = 0, b = NBINS - 1;
        for (; b > 0; --b) {
            int c = hist[b];
            if (cum + c >= CAP) break;
            cum += c;
        }
        s_bin = b; s_above = cum; ctrA = 0; ctrT = 0;
    }
    __syncthreads();

    int thr = s_bin, above = s_above, need = CAP - s_above;
    for (int i = tid; i < N_TOK; i += 256) {
        float p = probs[(size_t)i * E_NUM + e];
        unsigned b = __float_as_uint(p) >> 19;
        if (b > NBINS - 1) b = NBINS - 1;
        int slot = -1;
        if ((int)b > thr) {
            slot = atomicAdd(&ctrA, 1);
        } else if ((int)b == thr) {
            int tr = atomicAdd(&ctrT, 1);
            if (tr < need) slot = above + tr;
        }
        if (slot >= 0) {
            tIdx[(size_t)e * CAP + slot] = i;
            tPrb[(size_t)e * CAP + slot] = p;
        }
    }
}

// ---------------- gather selected token rows -> bf16 ----------------
__global__ __launch_bounds__(256)
void k_gather(const float* __restrict__ x, const int* __restrict__ tIdx, bf16* __restrict__ xt) {
    int bid = blockIdx.x;                  // e*CAP + c
    int tok = tIdx[bid];
    #pragma unroll
    for (int j = 0; j < 4; ++j) {
        int h = threadIdx.x + j * 256;
        xt[(size_t)bid * H_DIM + h] = (bf16)safe_clampf(x[(size_t)tok * H_DIM + h], 1000.0f);
    }
}

// ---------------- convert weights fp32 -> bf16 ----------------
__global__ void k_convw(const float* __restrict__ wg, const float* __restrict__ wu,
                        const float* __restrict__ wd,
                        bf16* __restrict__ og, bf16* __restrict__ ou, bf16* __restrict__ od) {
    size_t i = (size_t)blockIdx.x * blockDim.x + threadIdx.x;
    size_t st = (size_t)gridDim.x * blockDim.x;
    const size_t n = (size_t)E_NUM * I_DIM * H_DIM;
    for (; i < n; i += st) {
        __builtin_prefetch(&wg[i + st], 0, 1);   // global_prefetch_b8
        og[i] = (bf16)wg[i];
        ou[i] = (bf16)wu[i];
        od[i] = (bf16)wd[i];
    }
}

// ---------------- WMMA fragment loaders (16-bit A 16x32 / B 32x16 layouts) ----------------
union FragU { v16bf v; v8bf h[2]; };

// A-matrix 16x32: lane<16 row=lane K={0..7,16..23}; lane>=16 same row K={8..15,24..31}
__device__ __forceinline__ v16bf ld_frag_a(const bf16* tile, int lane) {
    int r = lane & 15, hs = lane >> 4;
    const bf16* p = tile + r * 32 + hs * 8;
    FragU u;
    u.h[0] = *(const v8bf*)p;
    u.h[1] = *(const v8bf*)(p + 16);
    return u.v;
}
// B-matrix 32x16: lane n holds column n; lanes<16 K={0..7,16..23}, lanes>=16 K={8..15,24..31}
__device__ __forceinline__ v16bf ld_frag_b(const bf16* tile, int lane) {
    int r = lane & 15, hs = lane >> 4;
    const bf16* p = tile + r * 32 + hs * 8;
    FragU u;
    u.h[0] = *(const v8bf*)p;
    u.h[1] = *(const v8bf*)(p + 16);
    return u.v;
}

// ---------------- fused gate/up GEMM, async double-buffered LDS staging ----------------
// grid: (I/64, CAP/128, E), 256 threads = 8 waves, each wave 32x32 per matrix
__global__ __launch_bounds__(256)
void k_gemm_gateup(const bf16* __restrict__ xt, const bf16* __restrict__ wg,
                   const bf16* __restrict__ wu, bf16* __restrict__ act) {
    __shared__ bf16 lA[2][128 * 32];
    __shared__ bf16 lBg[2][64 * 32];
    __shared__ bf16 lBu[2][64 * 32];

    int e = blockIdx.z;
    int n0 = blockIdx.x * 64, m0 = blockIdx.y * 128;
    const bf16* A  = xt + (size_t)e * CAP * H_DIM;
    const bf16* Bg = wg + (size_t)e * I_DIM * H_DIM;
    const bf16* Bu = wu + (size_t)e * I_DIM * H_DIM;

    int tid = threadIdx.x, lane = tid & 31, wid = tid >> 5;
    int wm = (wid & 3) * 32, wn = (wid >> 2) * 32;

    // per-thread staging coordinates: A 16 halves (2x b128), Bg/Bu 8 halves (1x b128 each)
    int ra = (tid * 16) >> 5, ca = (tid * 16) & 31;
    int rb = (tid * 8) >> 5,  cb = (tid * 8) & 31;

    v8f ag[2][2], au[2][2];
    #pragma unroll
    for (int a = 0; a < 2; ++a)
        #pragma unroll
        for (int b = 0; b < 2; ++b)
            #pragma unroll
            for (int k = 0; k < 8; ++k) { ag[a][b][k] = 0.0f; au[a][b][k] = 0.0f; }

    // prologue: stage slab 0 into buffer 0 (4 async instrs per wave)
    async_b128(lds_off(&lA[0][ra * 32 + ca]),     A + (size_t)(m0 + ra) * H_DIM + ca);
    async_b128(lds_off(&lA[0][ra * 32 + ca + 8]), A + (size_t)(m0 + ra) * H_DIM + ca + 8);
    async_b128(lds_off(&lBg[0][rb * 32 + cb]),    Bg + (size_t)(n0 + rb) * H_DIM + cb);
    async_b128(lds_off(&lBu[0][rb * 32 + cb]),    Bu + (size_t)(n0 + rb) * H_DIM + cb);

    const int NK = H_DIM / 32;
    for (int it = 0; it < NK; ++it) {
        int cur = it & 1, nxt = cur ^ 1;
        if (it + 1 < NK) {   // prefetch next slab into other buffer, then wait on current
            int k0 = (it + 1) * 32;
            async_b128(lds_off(&lA[nxt][ra * 32 + ca]),     A + (size_t)(m0 + ra) * H_DIM + k0 + ca);
            async_b128(lds_off(&lA[nxt][ra * 32 + ca + 8]), A + (size_t)(m0 + ra) * H_DIM + k0 + ca + 8);
            async_b128(lds_off(&lBg[nxt][rb * 32 + cb]),    Bg + (size_t)(n0 + rb) * H_DIM + k0 + cb);
            async_b128(lds_off(&lBu[nxt][rb * 32 + cb]),    Bu + (size_t)(n0 + rb) * H_DIM + k0 + cb);
            WAIT_ASYNC(4);   // current slab's 4 per-wave async ops complete (in-order)
        } else {
            WAIT_ASYNC(0);
        }
        __syncthreads();

        v16bf af[2], bgf[2], buf2[2];
        #pragma unroll
        for (int mi = 0; mi < 2; ++mi) af[mi] = ld_frag_a(&lA[cur][(wm + mi * 16) * 32], lane);
        #pragma unroll
        for (int ni = 0; ni < 2; ++ni) {
            bgf[ni]  = ld_frag_b(&lBg[cur][(wn + ni * 16) * 32], lane);
            buf2[ni] = ld_frag_b(&lBu[cur][(wn + ni * 16) * 32], lane);
        }
        #pragma unroll
        for (int mi = 0; mi < 2; ++mi)
            #pragma unroll
            for (int ni = 0; ni < 2; ++ni) {
                ag[mi][ni] = __builtin_amdgcn_wmma_f32_16x16x32_bf16(
                    false, af[mi], false, bgf[ni], (short)0, ag[mi][ni], false, false);
                au[mi][ni] = __builtin_amdgcn_wmma_f32_16x16x32_bf16(
                    false, af[mi], false, buf2[ni], (short)0, au[mi][ni], false, false);
            }
        __syncthreads();   // all waves done reading before buffer is refilled
    }

    // epilogue: C/D layout: VGPR r -> M = r + 8*(lane>=16), N = lane&15
    #pragma unroll
    for (int mi = 0; mi < 2; ++mi)
        #pragma unroll
        for (int ni = 0; ni < 2; ++ni)
            #pragma unroll
            for (int r = 0; r < 8; ++r) {
                int c = m0 + wm + mi * 16 + r + ((lane >> 4) << 3);
                int i = n0 + wn + ni * 16 + (lane & 15);
                float g = safe_clampf(siluf(ag[mi][ni][r]), 1000.0f);
                float u = safe_clampf(au[mi][ni][r], 1000.0f);
                act[(size_t)e * CAP * I_DIM + (size_t)c * I_DIM + i] = (bf16)(g * u);
            }
}

// ---------------- down GEMM (double-buffered TDM tiles) + weighted scatter-add ----------------
// grid: (H/64, CAP/128, E)
__global__ __launch_bounds__(256)
void k_gemm_down(const bf16* __restrict__ act, const bf16* __restrict__ wd,
                 const int* __restrict__ tIdx, const float* __restrict__ tPrb,
                 float* __restrict__ outacc) {
    __shared__ bf16 lA[2][128 * 32];
    __shared__ bf16 lB[2][64 * 32];

    int e = blockIdx.z;
    int h0 = blockIdx.x * 64, m0 = blockIdx.y * 128;
    const bf16* A = act + (size_t)e * CAP * I_DIM;
    const bf16* B = wd + (size_t)e * H_DIM * I_DIM;

    int tid = threadIdx.x, lane = tid & 31, wid = tid >> 5;
    int wm = (wid & 3) * 32, wn = (wid >> 2) * 32;

    v8f acc[2][2];
    #pragma unroll
    for (int a = 0; a < 2; ++a)
        #pragma unroll
        for (int b = 0; b < 2; ++b)
            #pragma unroll
            for (int k = 0; k < 8; ++k) acc[a][b][k] = 0.0f;

    const int NK = I_DIM / 32;
#if defined(HAVE_TDM)
    if (wid == 0) {   // prologue: TDM pair for slab 0 into buffer 0
        tdm_load_2d(lds_off(&lA[0][0]), A + (size_t)m0 * I_DIM,
                    /*tile*/32, 128, /*tensor*/I_DIM, CAP, /*stride0*/I_DIM);
        tdm_load_2d(lds_off(&lB[0][0]), B + (size_t)h0 * I_DIM,
                    /*tile*/32, 64, /*tensor*/I_DIM, H_DIM, /*stride0*/I_DIM);
    }
#endif
    for (int it = 0; it < NK; ++it) {
        int cur = it & 1, nxt = cur ^ 1;
#if defined(HAVE_TDM)
        if (wid == 0) {   // issue next slab, then wait so only the 2 just-issued remain
            if (it + 1 < NK) {
                int k0 = (it + 1) * 32;
                tdm_load_2d(lds_off(&lA[nxt][0]), A + (size_t)m0 * I_DIM + k0,
                            32, 128, I_DIM, CAP, I_DIM);
                tdm_load_2d(lds_off(&lB[nxt][0]), B + (size_t)h0 * I_DIM + k0,
                            32, 64, I_DIM, H_DIM, I_DIM);
                WAIT_TENSOR_N(2);   // TDM ops of one wave complete in order
            } else {
                WAIT_TENSOR_N(0);
            }
        }
        __syncthreads();
#else
        {
            int k0 = it * 32;
            int idx = tid * 16, r = idx >> 5, c = idx & 31;
            const bf16* src = A + (size_t)(m0 + r) * I_DIM + k0 + c;
            *(v8bf*)&lA[cur][r * 32 + c]     = *(const v8bf*)src;
            *(v8bf*)&lA[cur][r * 32 + c + 8] = *(const v8bf*)(src + 8);
            int idx2 = tid * 8, r2 = idx2 >> 5, c2 = idx2 & 31;
            *(v8bf*)&lB[cur][r2 * 32 + c2] = *(const v8bf*)(B + (size_t)(h0 + r2) * I_DIM + k0 + c2);
        }
        __syncthreads();
#endif
        v16bf af[2], bf[2];
        #pragma unroll
        for (int mi = 0; mi < 2; ++mi) af[mi] = ld_frag_a(&lA[cur][(wm + mi * 16) * 32], lane);
        #pragma unroll
        for (int ni = 0; ni < 2; ++ni) bf[ni] = ld_frag_b(&lB[cur][(wn + ni * 16) * 32], lane);
        #pragma unroll
        for (int mi = 0; mi < 2; ++mi)
            #pragma unroll
            for (int ni = 0; ni < 2; ++ni)
                acc[mi][ni] = __builtin_amdgcn_wmma_f32_16x16x32_bf16(
                    false, af[mi], false, bf[ni], (short)0, acc[mi][ni], false, false);
        __syncthreads();   // all waves done reading cur before wave0 refills it
    }

    #pragma unroll
    for (int mi = 0; mi < 2; ++mi)
        #pragma unroll
        for (int ni = 0; ni < 2; ++ni)
            #pragma unroll
            for (int r = 0; r < 8; ++r) {
                int c = m0 + wm + mi * 16 + r + ((lane >> 4) << 3);
                int h = h0 + wn + ni * 16 + (lane & 15);
                int tok = tIdx[(size_t)e * CAP + c];
                float p = tPrb[(size_t)e * CAP + c];
                float v = safe_clampf(acc[mi][ni][r], 1000.0f);
                atomicAdd(&outacc[(size_t)tok * H_DIM + h], p * v);
            }
}

// ---------------- counts scatter ----------------
__global__ void k_counts(const int* __restrict__ tIdx, const float* __restrict__ tPrb,
                         float* __restrict__ counts) {
    int i = blockIdx.x * blockDim.x + threadIdx.x;
    if (i < E_NUM * CAP) atomicAdd(&counts[tIdx[i]], tPrb[i]);
}

// ---------------- finalize: divide by counts, clamp, aux ----------------
__global__ __launch_bounds__(256)
void k_final(float* __restrict__ out, const float* __restrict__ counts,
             const float* __restrict__ lse2) {
    int n = blockIdx.x, tid = threadIdx.x;
    float inv = 1.0f / fmaxf(counts[n], 1e-5f);
    #pragma unroll
    for (int j = 0; j < 4; ++j) {
        size_t idx = (size_t)n * H_DIM + tid + j * 256;
        out[idx] = safe_clampf(out[idx] * inv, 1000.0f);
    }
    if (n == 0 && tid == 0) {
        float s = lse2[0];
        float aux = s * (0.001f / (float)N_TOK);
        aux = fminf(fmaxf(aux, 0.0f), 10.0f);
        if (__builtin_isnan(aux) || __builtin_isinf(aux)) aux = 0.1f;
        out[(size_t)N_TOK * H_DIM] = aux;
    }
}

// ---------------- host launcher ----------------
extern "C" void kernel_launch(void* const* d_in, const int* in_sizes, int n_in,
                              void* d_out, int out_size, void* d_ws, size_t ws_size,
                              hipStream_t stream) {
    const float* hidden = (const float*)d_in[0];
    const float* gate_w = (const float*)d_in[1];
    const float* gate_proj = (const float*)d_in[2];
    const float* up_proj = (const float*)d_in[3];
    const float* down_proj = (const float*)d_in[4];
    float* out = (float*)d_out;

    // workspace layout (bytes); total ~203 MB
    char* base = (char*)d_ws;
    const size_t WN = (size_t)E_NUM * I_DIM * H_DIM;            // 16,777,216 weight elems
    size_t off = 0;
    float* lse2   = (float*)(base + off); off += 256;
    float* logits = (float*)(base + off); off += (size_t)N_TOK * E_NUM * 4;
    float* probs  = (float*)(base + off); off += (size_t)N_TOK * E_NUM * 4;
    int*   tIdx   = (int*)  (base + off); off += (size_t)E_NUM * CAP * 4;
    float* tPrb   = (float*)(base + off); off += (size_t)E_NUM * CAP * 4;
    float* counts = (float*)(base + off); off += (size_t)N_TOK * 4;
    off = (off + 255) & ~(size_t)255;
    bf16* xt  = (bf16*)(base + off); off += (size_t)E_NUM * CAP * H_DIM * 2;
    bf16* wgb = (bf16*)(base + off); off += WN * 2;
    bf16* wub = (bf16*)(base + off); off += WN * 2;
    bf16* wdb = (bf16*)(base + off); off += WN * 2;
    bf16* actb = (bf16*)(base + off); off += (size_t)E_NUM * CAP * I_DIM * 2;
    (void)ws_size; (void)n_in; (void)in_sizes; (void)out_size;

    k_zero<<<4096, 256, 0, stream>>>(out, N_TOK * H_DIM + 1);
    k_zero<<<64, 256, 0, stream>>>(counts, N_TOK);
    k_zero<<<1, 64, 0, stream>>>(lse2, 1);

    k_ln_gate<<<N_TOK, 256, 0, stream>>>(hidden, gate_w, logits, lse2);
    k_softmax<<<E_NUM, 1024, 0, stream>>>(logits, probs);
    k_topk<<<E_NUM, 256, 0, stream>>>(probs, tIdx, tPrb);
    k_gather<<<E_NUM * CAP, 256, 0, stream>>>(hidden, tIdx, xt);
    k_convw<<<4096, 256, 0, stream>>>(gate_proj, up_proj, down_proj, wgb, wub, wdb);

    k_gemm_gateup<<<dim3(I_DIM / 64, CAP / 128, E_NUM), 256, 0, stream>>>(xt, wgb, wub, actb);
    k_gemm_down<<<dim3(H_DIM / 64, CAP / 128, E_NUM), 256, 0, stream>>>(actb, wdb, tIdx, tPrb, out);

    k_counts<<<(E_NUM * CAP) / 256, 256, 0, stream>>>(tIdx, tPrb, counts);
    k_final<<<N_TOK, 256, 0, stream>>>(out, counts, lse2);
}
